// Dev_14139032338941
// MI455X (gfx1250) — matrix operand order repaired and checked by reading, not compiled, not run on hardware
//
#include <hip/hip_runtime.h>
#include <hip/hip_bf16.h>

typedef __bf16 bf16;
typedef __attribute__((ext_vector_type(8)))  bf16  v8bf;
typedef __attribute__((ext_vector_type(16))) bf16  v16bf;
typedef __attribute__((ext_vector_type(8)))  float v8f;
typedef __attribute__((ext_vector_type(4)))  int   v4i;

#if defined(__has_builtin)
#if __has_builtin(__builtin_amdgcn_global_load_async_to_lds_b128)
#define HAVE_ASYNC_LDS 1
#else
#define HAVE_ASYNC_LDS 0
#endif
#else
#define HAVE_ASYNC_LDS 0
#endif

#if HAVE_ASYNC_LDS
typedef __attribute__((address_space(1))) v4i* gv4i_p;
typedef __attribute__((address_space(3))) v4i* lv4i_p;
#endif

// 16-byte global -> LDS copy; async DMA path on CDNA5 when available.
__device__ __forceinline__ void g2lds_b128(const bf16* g, bf16* l) {
#if HAVE_ASYNC_LDS
  __builtin_amdgcn_global_load_async_to_lds_b128(
      (gv4i_p)(unsigned long long)g,
      (lv4i_p)(unsigned)(unsigned long long)l, 0, 0);
#else
  *(v8bf*)l = *(const v8bf*)g;
#endif
}

__device__ __forceinline__ void async_wait() {
#if HAVE_ASYNC_LDS
#if __has_builtin(__builtin_amdgcn_s_wait_asynccnt)
  __builtin_amdgcn_s_wait_asynccnt(0);
#else
  asm volatile("s_wait_asynccnt 0x0" ::: "memory");
#endif
#endif
}

// ---------------------------------------------------------------------------
// WMMA fragment loaders (CDNA5 16x16x32 bf16 layouts, wave32)
// A (16x32, MxK): lane L (row M=L&15): L<16 -> K0..7,K16..23 ; L>=16 -> K8..15,K24..31
// B (32x16, KxN): lane L (col N=L&15): L<16 -> K0..7,K16..23 ; L>=16 -> K8..15,K24..31
// LDS tile: [16 rows/cols][32 K] bf16, 16B aligned -> two ds_load_b128 per frag.
// ---------------------------------------------------------------------------
__device__ __forceinline__ v16bf frag_a(const bf16* tile) {
  const int lane = threadIdx.x & 31;
  const int row  = lane & 15;
  const int ko   = (lane >> 4) << 3;           // 0 or 8
  v8bf lo = *(const v8bf*)(tile + row * 32 + ko);
  v8bf hi = *(const v8bf*)(tile + row * 32 + ko + 16);
  v16bf f;
#pragma unroll
  for (int i = 0; i < 8; ++i) { f[i] = lo[i]; f[i + 8] = hi[i]; }
  return f;
}

__device__ __forceinline__ v16bf frag_b(const bf16* tile) {
  const int lane = threadIdx.x & 31;
  const int col  = lane & 15;
  const int kb   = (lane >> 4) << 3;           // 0 or 8
  v8bf lo = *(const v8bf*)(tile + col * 32 + kb);
  v8bf hi = *(const v8bf*)(tile + col * 32 + kb + 16);
  v16bf f;
#pragma unroll
  for (int i = 0; i < 8; ++i) { f[i] = lo[i]; f[i + 8] = hi[i]; }
  return f;
}

__device__ __forceinline__ v8f wmma_bf16(v16bf a, v16bf b, v8f c) {
  return __builtin_amdgcn_wmma_f32_16x16x32_bf16(false, a, false, b,
                                                 (short)0, c, false, false);
}

__device__ __forceinline__ float bn_relu(float acc, int m,
    const float* __restrict__ cb, const float* __restrict__ g,
    const float* __restrict__ bb, const float* __restrict__ mm,
    const float* __restrict__ vv) {
  float s = g[m] * rsqrtf(vv[m] + 1e-5f);
  return fmaxf((acc + cb[m] - mm[m]) * s + bb[m], 0.0f);
}

// ---------------------------------------------------------------------------
// ROI level assignment:  lvl = clip(round(4 + log2(sqrt(h*w)/(224/512))), 2, 5)
// ---------------------------------------------------------------------------
__global__ void __launch_bounds__(256)
lvl_kernel(const float* __restrict__ rois, int* __restrict__ lvl, int nR) {
  int r = blockIdx.x * blockDim.x + threadIdx.x;
  if (r >= nR) return;
  float y1 = rois[r * 4 + 0], x1 = rois[r * 4 + 1];
  float y2 = rois[r * 4 + 2], x2 = rois[r * 4 + 3];
  float h = y2 - y1, w = x2 - x1;
  float l = 4.0f + log2f(sqrtf(h * w) / (224.0f / 512.0f));
  l = rintf(l);
  l = fminf(fmaxf(l, 2.0f), 5.0f);
  lvl[r] = (int)l;
}

// ---------------------------------------------------------------------------
// Weight pre-pack: convert f32 conv weights to bf16 in the exact LDS A-tile
// layout: [kt][mt][sub][row16][kl32], so GEMM A-fills are contiguous copies.
// mode 0: convT (phase-major, K/phase = 256,512,512,1024, M=256, Mt=4)
// mode 1: conv3x3 OIHW 256->256 (K=2304, Mt=4)
// mode 2: fe_c1 OIHW 256->512 (K=2304, Mt=8)
// mode 3: fe_c2 (1024 x 25088 row-major, Mt=16)
// ---------------------------------------------------------------------------
__global__ void __launch_bounds__(256)
pack_kernel(int mode, const float* __restrict__ w, bf16* __restrict__ dst,
            long long total) {
  long long idx = (long long)blockIdx.x * blockDim.x + threadIdx.x;
  if (idx >= total) return;
  float val;
  if (mode == 0) {
    int p = (idx >= 327680) ? 3 : (idx >= 196608) ? 2 : (idx >= 65536) ? 1 : 0;
    long long off = (p == 0) ? 0 : (p == 1) ? 65536 : (p == 2) ? 196608 : 327680;
    long long local = idx - off;
    int blockid = (int)(local >> 11);
    int within  = (int)(local & 2047);
    int mt = blockid & 3, kt = blockid >> 2;
    int sub = within >> 9, row = (within >> 5) & 15, kl = within & 31;
    int k = kt * 32 + kl;
    int m = mt * 64 + sub * 16 + row;
    int ci = k & 255, t = k >> 8;
    int py = p >> 1, px = p & 1;
    int ntx = px ? 2 : 1;
    int tyi = t / ntx, txi = t % ntx;
    int wy = py ? (2 - 2 * tyi) : 1;
    int wx = px ? (2 - 2 * txi) : 1;
    val = w[(((size_t)ci * 256 + m) * 3 + wy) * 3 + wx];
  } else if (mode == 1 || mode == 2) {
    int Mt = (mode == 1) ? 4 : 8;
    int blockid = (int)(idx >> 11);
    int within  = (int)(idx & 2047);
    int mt = blockid % Mt, kt = blockid / Mt;
    int sub = within >> 9, row = (within >> 5) & 15, kl = within & 31;
    int k = kt * 32 + kl;
    int m = mt * 64 + sub * 16 + row;
    int ci = k & 255, t = k >> 8;
    val = w[((size_t)m * 256 + ci) * 9 + (t / 3) * 3 + (t % 3)];
  } else {
    int blockid = (int)(idx >> 11);
    int within  = (int)(idx & 2047);
    int mt = blockid & 15, kt = blockid >> 4;
    int sub = within >> 9, row = (within >> 5) & 15, kl = within & 31;
    int k = kt * 32 + kl;
    int m = mt * 64 + sub * 16 + row;
    val = w[(size_t)m * 25088 + k];
  }
  dst[idx] = (bf16)val;
}

// ---------------------------------------------------------------------------
// Transposed conv 3x3 stride2 (subpixel phases) + BN + ReLU.
// in: (2,256,H,W) f32, out: (2,256,2H,2W) f32.  blockIdx.z = b*4+phase.
// GEMM per phase: M=256, N=H*W, K=ntaps*256. Block tile 64x128, 2x2 accs/wave.
// A comes pre-packed (packA + phase offset); fill = contiguous async copy.
// ---------------------------------------------------------------------------
__global__ void __launch_bounds__(256)
convt_phase_kernel(const float* __restrict__ in, float* __restrict__ out,
                   const bf16* __restrict__ packA, const float* __restrict__ cb,
                   const float* __restrict__ g, const float* __restrict__ bb,
                   const float* __restrict__ mm, const float* __restrict__ vv,
                   int H, int W) {
  __shared__ __align__(16) bf16 As[2][4][16][32];
  __shared__ __align__(16) bf16 Bs[2][8][16][32];

  const int tid = threadIdx.x;
  const int warp = tid >> 5, lane = tid & 31;
  const int wm = warp >> 2, wn = warp & 3;
  const int bz = blockIdx.z, b = bz >> 2, ph = bz & 3;
  const int py = ph >> 1, px = ph & 1;
  const int ntx = px ? 2 : 1;
  const int nty = py ? 2 : 1;
  const int K = (nty * ntx) << 8;
  const int mtile = blockIdx.y;
  const int tileN = blockIdx.x * 128, tileM = mtile * 64;
  const size_t HW = (size_t)H * W;
  const float* inb = in + (size_t)b * 256 * HW;
  const int ctoff = (ph == 0) ? 0 : (ph == 1) ? 65536 : (ph == 2) ? 196608 : 327680;
  const bf16* pA = packA + ctoff;

  const int bcol = tid >> 1;              // 0..127
  const int bkc  = (tid & 1) * 16;        // K-chunk of 16
  const int bn   = tileN + bcol;
  const int by   = bn / W, bx = bn % W;

  auto fillB = [&](int pbuf, int k0) {
    const int kk  = k0 + bkc;
    const int t   = kk >> 8;              // constant over the 16-chunk
    const int ci0 = kk & 255;
    const int tyi = t / ntx, txi = t % ntx;
    const int iy = by + tyi, ix = bx + txi;
    const bool ok = (iy < H) && (ix < W);
    const int iyc = iy < H ? iy : H - 1;
    const int ixc = ix < W ? ix : W - 1;
    const float* src = inb + (size_t)ci0 * HW + (size_t)iyc * W + ixc;
    v8bf o0, o1;
#pragma unroll
    for (int j = 0; j < 8; ++j) {
      float f0 = src[(size_t)j * HW];
      float f1 = src[(size_t)(j + 8) * HW];
      o0[j] = (bf16)(ok ? f0 : 0.0f);
      o1[j] = (bf16)(ok ? f1 : 0.0f);
    }
    *(v8bf*)&Bs[pbuf][bcol >> 4][bcol & 15][bkc]     = o0;
    *(v8bf*)&Bs[pbuf][bcol >> 4][bcol & 15][bkc + 8] = o1;
  };
  auto fillA = [&](int pbuf, int k0) {
    const bf16* src = pA + (((size_t)(k0 >> 5) * 4) + mtile) * 2048 + tid * 8;
    g2lds_b128(src, &As[pbuf][0][0][0] + tid * 8);
  };

  v8f c00 = {0.f,0.f,0.f,0.f,0.f,0.f,0.f,0.f};
  v8f c01 = c00, c10 = c00, c11 = c00;

  fillA(0, 0); fillB(0, 0);
  async_wait();
  __syncthreads();
  int pb = 0;
  for (int k0 = 0; k0 < K; k0 += 32) {
    if (k0 + 32 < K) { fillA(pb ^ 1, k0 + 32); fillB(pb ^ 1, k0 + 32); }
    v16bf a0 = frag_a(&As[pb][wm * 2 + 0][0][0]);
    v16bf a1 = frag_a(&As[pb][wm * 2 + 1][0][0]);
    v16bf b0 = frag_b(&Bs[pb][wn * 2 + 0][0][0]);
    v16bf b1 = frag_b(&Bs[pb][wn * 2 + 1][0][0]);
    c00 = wmma_bf16(a0, b0, c00);
    c01 = wmma_bf16(a0, b1, c01);
    c10 = wmma_bf16(a1, b0, c10);
    c11 = wmma_bf16(a1, b1, c11);
    async_wait();
    __syncthreads();
    pb ^= 1;
  }

  const int OH = 2 * H, OW = 2 * W;
  float* outb = out + (size_t)b * 256 * OH * OW;
  const int lm = (lane >> 4) * 8, ln = lane & 15;
  const int m0 = tileM + (wm * 2 + 0) * 16 + lm;
  const int m1 = tileM + (wm * 2 + 1) * 16 + lm;
  const int n0 = tileN + (wn * 2 + 0) * 16 + ln;
  const int n1 = tileN + (wn * 2 + 1) * 16 + ln;

  auto epi = [&](v8f c, int mb, int n) {
    int y = n / W, x = n % W;
    int oy = 2 * y + py, ox = 2 * x + px;
#pragma unroll
    for (int i = 0; i < 8; ++i) {
      int m = mb + i;
      outb[((size_t)m * OH + oy) * OW + ox] = bn_relu(c[i], m, cb, g, bb, mm, vv);
    }
  };
  epi(c00, m0, n0); epi(c01, m0, n1); epi(c10, m1, n0); epi(c11, m1, n1);
}

// ---------------------------------------------------------------------------
// Conv 3x3 stride1 pad1 (256->256) + BN + ReLU, f32 in/out.
// GEMM: M=256, N=H*W, K=2304. Block tile 64x128, 2x2 accs/wave. Packed A.
// ---------------------------------------------------------------------------
__global__ void __launch_bounds__(256)
conv3x3_kernel(const float* __restrict__ in, float* __restrict__ out,
               const bf16* __restrict__ packA, const float* __restrict__ cb,
               const float* __restrict__ g, const float* __restrict__ bb,
               const float* __restrict__ mm, const float* __restrict__ vv,
               int H, int W) {
  __shared__ __align__(16) bf16 As[2][4][16][32];
  __shared__ __align__(16) bf16 Bs[2][8][16][32];

  const int tid = threadIdx.x;
  const int warp = tid >> 5, lane = tid & 31;
  const int wm = warp >> 2, wn = warp & 3;
  const int b = blockIdx.z;
  const int mtile = blockIdx.y;
  const int tileN = blockIdx.x * 128, tileM = mtile * 64;
  const size_t HW = (size_t)H * W;
  const float* inb = in + (size_t)b * 256 * HW;

  const int bcol = tid >> 1;
  const int bkc  = (tid & 1) * 16;
  const int bn   = tileN + bcol;
  const int by   = bn / W, bx = bn % W;

  auto fillB = [&](int pbuf, int k0) {
    const int kk  = k0 + bkc;
    const int t   = kk >> 8;
    const int ci0 = kk & 255;
    const int ky = t / 3, kx = t % 3;
    const int iy = by - 1 + ky, ix = bx - 1 + kx;
    const bool ok = (iy >= 0) && (iy < H) && (ix >= 0) && (ix < W);
    const int iyc = iy < 0 ? 0 : (iy >= H ? H - 1 : iy);
    const int ixc = ix < 0 ? 0 : (ix >= W ? W - 1 : ix);
    const float* src = inb + (size_t)ci0 * HW + (size_t)iyc * W + ixc;
    v8bf o0, o1;
#pragma unroll
    for (int j = 0; j < 8; ++j) {
      float f0 = src[(size_t)j * HW];
      float f1 = src[(size_t)(j + 8) * HW];
      o0[j] = (bf16)(ok ? f0 : 0.0f);
      o1[j] = (bf16)(ok ? f1 : 0.0f);
    }
    *(v8bf*)&Bs[pbuf][bcol >> 4][bcol & 15][bkc]     = o0;
    *(v8bf*)&Bs[pbuf][bcol >> 4][bcol & 15][bkc + 8] = o1;
  };
  auto fillA = [&](int pbuf, int k0) {
    const bf16* src = packA + (((size_t)(k0 >> 5) * 4) + mtile) * 2048 + tid * 8;
    g2lds_b128(src, &As[pbuf][0][0][0] + tid * 8);
  };

  v8f c00 = {0.f,0.f,0.f,0.f,0.f,0.f,0.f,0.f};
  v8f c01 = c00, c10 = c00, c11 = c00;

  fillA(0, 0); fillB(0, 0);
  async_wait();
  __syncthreads();
  int pb = 0;
  for (int k0 = 0; k0 < 2304; k0 += 32) {
    if (k0 + 32 < 2304) { fillA(pb ^ 1, k0 + 32); fillB(pb ^ 1, k0 + 32); }
    v16bf a0 = frag_a(&As[pb][wm * 2 + 0][0][0]);
    v16bf a1 = frag_a(&As[pb][wm * 2 + 1][0][0]);
    v16bf b0 = frag_b(&Bs[pb][wn * 2 + 0][0][0]);
    v16bf b1 = frag_b(&Bs[pb][wn * 2 + 1][0][0]);
    c00 = wmma_bf16(a0, b0, c00);
    c01 = wmma_bf16(a0, b1, c01);
    c10 = wmma_bf16(a1, b0, c10);
    c11 = wmma_bf16(a1, b1, c11);
    async_wait();
    __syncthreads();
    pb ^= 1;
  }

  float* outb = out + (size_t)b * 256 * HW;
  const int lm = (lane >> 4) * 8, ln = lane & 15;
  const int m0 = tileM + (wm * 2 + 0) * 16 + lm;
  const int m1 = tileM + (wm * 2 + 1) * 16 + lm;
  const int n0 = tileN + (wn * 2 + 0) * 16 + ln;
  const int n1 = tileN + (wn * 2 + 1) * 16 + ln;

  auto epi = [&](v8f c, int mb, int n) {
    int y = n / W, x = n % W;
#pragma unroll
    for (int i = 0; i < 8; ++i) {
      int m = mb + i;
      outb[((size_t)m * H + y) * W + x] = bn_relu(c[i], m, cb, g, bb, mm, vv);
    }
  };
  epi(c00, m0, n0); epi(c01, m0, n1); epi(c10, m1, n0); epi(c11, m1, n1);
}

// ---------------------------------------------------------------------------
// Bilinear crop-resize (mirrors reference _crop_resize incl. pre-clip mask).
// mode 0: per-ROI level select {up2,up3,x4,x5}, f32 out (pooled/maskp)
// mode 1: per-ROI level select, only lvl 2/3, bf16 out, zeros otherwise (small)
// mode 2: fixed feature map (x2 or x3), bf16 out (big2/big3 inputs)
// ---------------------------------------------------------------------------
__global__ void __launch_bounds__(256)
crop_kernel(int mode, int crop, float* __restrict__ outF, bf16* __restrict__ outB,
            const float* __restrict__ rois, const int* __restrict__ lvl,
            const float* __restrict__ up2, const float* __restrict__ up3,
            const float* __restrict__ x4, const float* __restrict__ x5,
            const float* __restrict__ fmfix, int Hfix, long long total) {
  long long idx = (long long)blockIdx.x * blockDim.x + threadIdx.x;
  if (idx >= total) return;
  const int cc = crop * crop;
  const int x = (int)(idx % crop);
  const int y = (int)((idx / crop) % crop);
  const int c = (int)((idx / cc) % 256);
  const int r = (int)(idx / ((long long)cc * 256));
  const int b = r / 200;

  const float* fm;
  int H;
  float s = 1.0f;
  if (mode == 2) { fm = fmfix; H = Hfix; }
  else {
    int L = lvl[r];
    if (L == 2)      { fm = up2; H = 256; s = 2.0f; }
    else if (L == 3) { fm = up3; H = 128; s = 2.0f; }
    else if (mode == 0) {
      if (L == 4) { fm = x4; H = 32; } else { fm = x5; H = 16; }
    } else { outB[idx] = (bf16)0.0f; return; }
  }

  float y1 = rois[r * 4 + 0] * s, xx1 = rois[r * 4 + 1] * s;
  float y2 = rois[r * 4 + 2] * s, xx2 = rois[r * 4 + 3] * s;
  float cm1 = (float)(H - 1);
  float iy = y1 * cm1 + (float)y * ((y2 - y1) * cm1 / (float)(crop - 1));
  float ix = xx1 * cm1 + (float)x * ((xx2 - xx1) * cm1 / (float)(crop - 1));
  bool valid = (iy >= 0.f) && (iy <= cm1) && (ix >= 0.f) && (ix <= cm1);
  iy = fminf(fmaxf(iy, 0.f), cm1);
  ix = fminf(fmaxf(ix, 0.f), cm1);
  float fy0 = floorf(iy), fx0 = floorf(ix);
  float ly = iy - fy0, lx = ix - fx0;
  int y0i = (int)fy0, y1i = (int)ceilf(iy);
  int x0i = (int)fx0, x1i = (int)ceilf(ix);
  const float* base = fm + ((size_t)b * 256 + c) * H * H;
  float tl = base[y0i * H + x0i], tr = base[y0i * H + x1i];
  float bl = base[y1i * H + x0i], br = base[y1i * H + x1i];
  float top = tl + (tr - tl) * lx;
  float bot = bl + (br - bl) * lx;
  float val = valid ? (top + (bot - top) * ly) : 0.0f;
  if (mode == 0) outF[idx] = val;
  else           outB[idx] = (bf16)val;
}

// ---------------------------------------------------------------------------
// fe conv1: 3x3 stride2 pad1 (256->512) on 14x14 bf16 crops -> 7x7, BN+ReLU,
// act1[crop][512][49] bf16. Block tile 64(M) x 64(N=49 valid), 2 accs/wave.
// Grid: (1200, 8). K = 2304. Packed A (Mt=8).
// ---------------------------------------------------------------------------
__global__ void __launch_bounds__(256)
fe_conv1_kernel(const bf16* __restrict__ crops, bf16* __restrict__ act1,
                const bf16* __restrict__ packA, const float* __restrict__ cb,
                const float* __restrict__ g, const float* __restrict__ bb,
                const float* __restrict__ mm, const float* __restrict__ vv) {
  __shared__ __align__(16) bf16 As[2][4][16][32];
  __shared__ __align__(16) bf16 Bs[2][4][16][32];

  const int tid = threadIdx.x;
  const int warp = tid >> 5, lane = tid & 31;
  const int wm = warp >> 2, wn = warp & 3;
  const int cropi = blockIdx.x;
  const int mtile = blockIdx.y;
  const int tileM = mtile * 64;
  const bf16* cin = crops + (size_t)cropi * 256 * 196;

  const int bcol = tid >> 2;              // 0..63 (output site)
  const int bkc  = (tid & 3) * 8;
  const int oy = bcol / 7, ox = bcol % 7; // valid when bcol < 49

  auto fillB = [&](int pbuf, int k0) {
    const int kk  = k0 + bkc;
    const int t   = kk >> 8;
    const int ci0 = kk & 255;
    const int ky = t / 3, kx = t % 3;
    const int iy = 2 * oy - 1 + ky, ix = 2 * ox - 1 + kx;
    const bool ok = (bcol < 49) && (iy >= 0) && (iy < 14) && (ix >= 0) && (ix < 14);
    const int iyc = iy < 0 ? 0 : (iy > 13 ? 13 : iy);
    const int ixc = ix < 0 ? 0 : (ix > 13 ? 13 : ix);
    const bf16* src = cin + (size_t)ci0 * 196 + iyc * 14 + ixc;
    v8bf o;
#pragma unroll
    for (int j = 0; j < 8; ++j) {
      bf16 v = src[(size_t)j * 196];
      o[j] = ok ? v : (bf16)0.0f;
    }
    *(v8bf*)&Bs[pbuf][bcol >> 4][bcol & 15][bkc] = o;
  };
  auto fillA = [&](int pbuf, int k0) {
    const bf16* src = packA + (((size_t)(k0 >> 5) * 8) + mtile) * 2048 + tid * 8;
    g2lds_b128(src, &As[pbuf][0][0][0] + tid * 8);
  };

  v8f c0 = {0.f,0.f,0.f,0.f,0.f,0.f,0.f,0.f};
  v8f c1 = c0;

  fillA(0, 0); fillB(0, 0);
  async_wait();
  __syncthreads();
  int pb = 0;
  for (int k0 = 0; k0 < 2304; k0 += 32) {
    if (k0 + 32 < 2304) { fillA(pb ^ 1, k0 + 32); fillB(pb ^ 1, k0 + 32); }
    v16bf a0 = frag_a(&As[pb][wm * 2 + 0][0][0]);
    v16bf a1 = frag_a(&As[pb][wm * 2 + 1][0][0]);
    v16bf b0 = frag_b(&Bs[pb][wn][0][0]);
    c0 = wmma_bf16(a0, b0, c0);
    c1 = wmma_bf16(a1, b0, c1);
    async_wait();
    __syncthreads();
    pb ^= 1;
  }

  const int lm = (lane >> 4) * 8, ln = lane & 15;
  const int n = wn * 16 + ln;
  if (n >= 49) return;
  const int m0 = tileM + (wm * 2 + 0) * 16 + lm;
  const int m1 = tileM + (wm * 2 + 1) * 16 + lm;
#pragma unroll
  for (int i = 0; i < 8; ++i) {
    int m = m0 + i;
    act1[((size_t)cropi * 512 + m) * 49 + n] =
        (bf16)bn_relu(c0[i], m, cb, g, bb, mm, vv);
  }
#pragma unroll
  for (int i = 0; i < 8; ++i) {
    int m = m1 + i;
    act1[((size_t)cropi * 512 + m) * 49 + n] =
        (bf16)bn_relu(c1[i], m, cb, g, bb, mm, vv);
  }
}

// ---------------------------------------------------------------------------
// fe conv2: GEMM M=1024, K=25088, N=1200 crops, block tile 64x128,
// 2x2 accs/wave, double-buffered. A packed (Mt=16); B contiguous bf16.
// Both operands stream global->LDS via async b128 copies when available.
// Epilogue: BN+ReLU, level masks, scatter to small/big2/big3 segments.
// ---------------------------------------------------------------------------
__global__ void __launch_bounds__(256)
fe_conv2_kernel(const bf16* __restrict__ act1, const bf16* __restrict__ packA,
                const float* __restrict__ cb, const float* __restrict__ g,
                const float* __restrict__ bb, const float* __restrict__ mm,
                const float* __restrict__ vv, const int* __restrict__ lvl,
                float* __restrict__ o_small, float* __restrict__ o_big2,
                float* __restrict__ o_big3) {
  __shared__ __align__(16) bf16 As[2][4][16][32];
  __shared__ __align__(16) bf16 Bs[2][8][16][32];

  const int KK = 25088;
  const int tid = threadIdx.x;
  const int warp = tid >> 5, lane = tid & 31;
  const int wm = warp >> 2, wn = warp & 3;
  const int mtile = blockIdx.y;
  const int tileN = blockIdx.x * 128, tileM = mtile * 64;

  const int bcol = tid >> 1;
  const int bkc  = (tid & 1) * 16;
  const int nB   = tileN + bcol;

  auto fillB = [&](int pbuf, int k0) {
    bf16* dst = &Bs[pbuf][bcol >> 4][bcol & 15][bkc];
    if (nB < 1200) {
      const bf16* src = act1 + (size_t)nB * KK + k0 + bkc;
      g2lds_b128(src, dst);
      g2lds_b128(src + 8, dst + 8);
    } else {
      v8bf z;
#pragma unroll
      for (int j = 0; j < 8; ++j) z[j] = (bf16)0.0f;
      *(v8bf*)dst = z;
      *(v8bf*)(dst + 8) = z;
    }
  };
  auto fillA = [&](int pbuf, int k0) {
    const bf16* src = packA + (((size_t)(k0 >> 5) * 16) + mtile) * 2048 + tid * 8;
    g2lds_b128(src, &As[pbuf][0][0][0] + tid * 8);
  };

  v8f c00 = {0.f,0.f,0.f,0.f,0.f,0.f,0.f,0.f};
  v8f c01 = c00, c10 = c00, c11 = c00;

  fillA(0, 0); fillB(0, 0);
  async_wait();
  __syncthreads();
  int pb = 0;
  for (int k0 = 0; k0 < KK; k0 += 32) {
    if (k0 + 32 < KK) { fillA(pb ^ 1, k0 + 32); fillB(pb ^ 1, k0 + 32); }
    v16bf a0 = frag_a(&As[pb][wm * 2 + 0][0][0]);
    v16bf a1 = frag_a(&As[pb][wm * 2 + 1][0][0]);
    v16bf b0 = frag_b(&Bs[pb][wn * 2 + 0][0][0]);
    v16bf b1 = frag_b(&Bs[pb][wn * 2 + 1][0][0]);
    c00 = wmma_bf16(a0, b0, c00);
    c01 = wmma_bf16(a0, b1, c01);
    c10 = wmma_bf16(a1, b0, c10);
    c11 = wmma_bf16(a1, b1, c11);
    async_wait();
    __syncthreads();
    pb ^= 1;
  }

  const int lm = (lane >> 4) * 8, ln = lane & 15;
  const int m0 = tileM + (wm * 2 + 0) * 16 + lm;
  const int m1 = tileM + (wm * 2 + 1) * 16 + lm;
  const int n0 = tileN + (wn * 2 + 0) * 16 + ln;
  const int n1 = tileN + (wn * 2 + 1) * 16 + ln;

  auto epi = [&](v8f c, int mb, int n) {
    if (n >= 1200) return;
    const int set = n / 400;
    const int r = n - set * 400;
    const int L = lvl[r];
    const bool ok = (set == 0) ? (L == 2 || L == 3)
                   : (set == 1) ? (L >= 4) : (L == 5);
    float* dst = (set == 0) ? o_small : (set == 1) ? o_big2 : o_big3;
#pragma unroll
    for (int i = 0; i < 8; ++i) {
      int m = mb + i;
      float o = ok ? bn_relu(c[i], m, cb, g, bb, mm, vv) : 0.0f;
      dst[(size_t)r * 1024 + m] = o;
    }
  };
  epi(c00, m0, n0); epi(c01, m0, n1); epi(c10, m1, n0); epi(c11, m1, n1);
}

// ---------------------------------------------------------------------------
// Host-side orchestration
// ---------------------------------------------------------------------------
extern "C" void kernel_launch(void* const* d_in, const int* in_sizes, int n_in,
                              void* d_out, int out_size, void* d_ws, size_t ws_size,
                              hipStream_t stream) {
  (void)in_sizes; (void)n_in; (void)out_size; (void)ws_size;

  const float* x2   = (const float*)d_in[0];   // (2,256,128,128)
  const float* x3   = (const float*)d_in[1];   // (2,256,64,64)
  const float* x4   = (const float*)d_in[2];   // (2,256,32,32)
  const float* x5   = (const float*)d_in[3];   // (2,256,16,16)
  const float* rois = (const float*)d_in[4];   // (2,200,4)
  // d_in[5]: roi_cls_gt (unused)
  const float* ct_w = (const float*)d_in[6];   // (256,256,3,3) (Cin,Cout,kh,kw)
  const float* ct_b = (const float*)d_in[7];
  const float* b1g = (const float*)d_in[8],  *b1b = (const float*)d_in[9];
  const float* b1m = (const float*)d_in[10], *b1v = (const float*)d_in[11];
  const float* c_w = (const float*)d_in[12];   // (256,256,3,3) OIHW
  const float* c_b = (const float*)d_in[13];
  const float* b2g = (const float*)d_in[14], *b2b = (const float*)d_in[15];
  const float* b2m = (const float*)d_in[16], *b2v = (const float*)d_in[17];
  const float* f1w = (const float*)d_in[18];   // (512,256,3,3)
  const float* f1b = (const float*)d_in[19];
  const float* f1g = (const float*)d_in[20], *f1bb = (const float*)d_in[21];
  const float* f1m = (const float*)d_in[22], *f1v = (const float*)d_in[23];
  const float* f2w = (const float*)d_in[24];   // (1024,512,7,7) = 1024 x 25088
  const float* f2b = (const float*)d_in[25];
  const float* f2g = (const float*)d_in[26], *f2bb = (const float*)d_in[27];
  const float* f2m = (const float*)d_in[28], *f2v = (const float*)d_in[29];

  // output segments: pooled, maskp, big2, big3, small
  float* outF = (float*)d_out;
  const long long PO = 400ll * 256 * 49;
  const long long MO = 400ll * 256 * 196;
  float* o_pool  = outF;
  float* o_mask  = outF + PO;
  float* o_big2  = o_mask + MO;
  float* o_big3  = o_big2 + 400ll * 1024;
  float* o_small = o_big3 + 400ll * 1024;

  // workspace partition
  float* bufT = (float*)d_ws;                  // stage1 temp: 2*256*256*256 f32
  float* up2  = bufT + 33554432ull;            // 2*256*256*256 f32
  float* up3  = up2  + 33554432ull;            // 2*256*128*128 f32
  bf16*  crops = (bf16*)(up3 + 8388608ull);    // 1200*256*14*14 bf16
  bf16*  act1  = crops + 60211200ull;          // 1200*512*49 bf16
  int*   lvl   = (int*)(act1 + 30105600ull);   // 400 int (pad to 512)
  bf16*  packCT = (bf16*)(lvl + 512);          // 2304*256 = 589824 bf16
  bf16*  packC3 = packCT + 589824ull;          // 589824
  bf16*  packF1 = packC3 + 589824ull;          // 2304*512 = 1179648
  bf16*  packF2 = packF1 + 1179648ull;         // 25088*1024 = 25690112

  dim3 blk(256);

  lvl_kernel<<<2, blk, 0, stream>>>(rois, lvl, 400);

  // one-shot weight packs (f32 -> bf16, LDS tile layout)
  pack_kernel<<<(589824 + 255) / 256, blk, 0, stream>>>(0, ct_w, packCT, 589824);
  pack_kernel<<<(589824 + 255) / 256, blk, 0, stream>>>(1, c_w, packC3, 589824);
  pack_kernel<<<(1179648 + 255) / 256, blk, 0, stream>>>(2, f1w, packF1, 1179648);
  pack_kernel<<<(25690112 + 255) / 256, blk, 0, stream>>>(3, f2w, packF2, 25690112);

  // upsample x2: convT(4 phases) -> bufT (2,256,256,256), conv3x3 -> up2
  convt_phase_kernel<<<dim3(128, 4, 8), blk, 0, stream>>>(
      x2, bufT, packCT, ct_b, b1g, b1b, b1m, b1v, 128, 128);
  conv3x3_kernel<<<dim3(512, 4, 2), blk, 0, stream>>>(
      bufT, up2, packC3, c_b, b2g, b2b, b2m, b2v, 256, 256);

  // upsample x3 (reuse bufT): -> (2,256,128,128) -> up3
  convt_phase_kernel<<<dim3(32, 4, 8), blk, 0, stream>>>(
      x3, bufT, packCT, ct_b, b1g, b1b, b1m, b1v, 64, 64);
  conv3x3_kernel<<<dim3(128, 4, 2), blk, 0, stream>>>(
      bufT, up3, packC3, c_b, b2g, b2b, b2m, b2v, 128, 128);

  // crops
  const long long tot7  = 400ll * 256 * 49;
  const long long tot14 = 400ll * 256 * 196;
  crop_kernel<<<(unsigned)((tot7 + 255) / 256), blk, 0, stream>>>(
      0, 7, o_pool, nullptr, rois, lvl, up2, up3, x4, x5, nullptr, 0, tot7);
  crop_kernel<<<(unsigned)((tot14 + 255) / 256), blk, 0, stream>>>(
      0, 14, o_mask, nullptr, rois, lvl, up2, up3, x4, x5, nullptr, 0, tot14);
  crop_kernel<<<(unsigned)((tot14 + 255) / 256), blk, 0, stream>>>(
      1, 14, nullptr, crops, rois, lvl, up2, up3, x4, x5, nullptr, 0, tot14);
  crop_kernel<<<(unsigned)((tot14 + 255) / 256), blk, 0, stream>>>(
      2, 14, nullptr, crops + tot14, rois, lvl, up2, up3, x4, x5, x2, 128, tot14);
  crop_kernel<<<(unsigned)((tot14 + 255) / 256), blk, 0, stream>>>(
      2, 14, nullptr, crops + 2 * tot14, rois, lvl, up2, up3, x4, x5, x3, 64, tot14);

  // feature-extract head on 1200 crops
  fe_conv1_kernel<<<dim3(1200, 8), blk, 0, stream>>>(
      crops, act1, packF1, f1b, f1g, f1bb, f1m, f1v);
  fe_conv2_kernel<<<dim3(10, 16), blk, 0, stream>>>(
      act1, packF2, f2b, f2g, f2bb, f2m, f2v, lvl, o_small, o_big2, o_big3);
}
